// Head_30477087932690
// MI455X (gfx1250) — matrix operand order repaired and checked by reading, not compiled, not run on hardware
//
#include <hip/hip_runtime.h>

typedef __attribute__((ext_vector_type(16))) __bf16 v16bf;
typedef __attribute__((ext_vector_type(8)))  __bf16 v8bf;
typedef __attribute__((ext_vector_type(8)))  float  v8f;
typedef __attribute__((ext_vector_type(4)))  float  v4f;

#define Bc 4
#define Tc 4096
#define Cc 1024
#define Hc 64
#define BT (Bc * Tc)
#define BTH (Bc * Tc * Hc)

// ---------------------------------------------------------------------------
// Kernel 0: Wq/Wk/Wv fp32 [C,H] -> bf16 W^T [3][H][C] (B-fragment friendly)
// ---------------------------------------------------------------------------
__global__ void wt_transpose_kernel(const float* __restrict__ Wq,
                                    const float* __restrict__ Wk,
                                    const float* __restrict__ Wv,
                                    __bf16* __restrict__ Wt) {
    int idx = blockIdx.x * blockDim.x + threadIdx.x;
    if (idx >= 3 * Cc * Hc) return;
    int w   = idx / (Cc * Hc);
    int rem = idx % (Cc * Hc);
    int c   = rem / Hc;
    int h   = rem % Hc;
    const float* W = (w == 0) ? Wq : ((w == 1) ? Wk : Wv);
    Wt[(size_t)w * Hc * Cc + (size_t)h * Cc + c] = (__bf16)W[(size_t)c * Hc + h];
}

// ---------------------------------------------------------------------------
// Kernel 1: fused QKV projection, bf16 WMMA, software-pipelined.
//   q,k stored bf16 [B*T][H]; v stored transposed bf16 [B][H][T].
// ---------------------------------------------------------------------------
__global__ __launch_bounds__(256) void qkv_proj_kernel(
        const float* __restrict__ x, const __bf16* __restrict__ Wt,
        __bf16* __restrict__ qb, __bf16* __restrict__ kb,
        __bf16* __restrict__ vt) {
    const int lane = threadIdx.x & 31;
    const int wave = threadIdx.x >> 5;
    const int nl   = lane & 15;
    const int half = lane >> 4;
    const int row0 = blockIdx.x * 128 + wave * 16;   // row in [0, B*T)

    // per-lane base pointers
    const float*  pxbase = x + (size_t)(row0 + nl) * Cc + half * 16;
    const __bf16* wbase  = Wt + (size_t)nl * Cc + half * 16;

    // B fragment for flat tile idx = mat*4+nt at K-offset kk
    auto ldB = [&](int idx, int kk) -> v16bf {
        int mat = idx >> 2, nt = idx & 3;
        return *(const v16bf*)(wbase + ((size_t)mat * Hc + nt * 16) * Cc + kk);
    };

    v8f acc[12] = {};   // idx = mat*4 + nt

    // ---- prologue: A raw floats for kk=0, first two B fragments ----
    v4f xa0 = *(const v4f*)(pxbase + 0);
    v4f xa1 = *(const v4f*)(pxbase + 4);
    v4f xa2 = *(const v4f*)(pxbase + 8);
    v4f xa3 = *(const v4f*)(pxbase + 12);
    v16bf bq0 = ldB(0, 0);
    v16bf bq1 = ldB(1, 0);

    for (int kk = 0; kk < Cc; kk += 32) {
        // convert this iteration's A fragment (per-lane K = off+0..15)
        v16bf a;
#pragma unroll
        for (int i = 0; i < 4; ++i) {
            a[i]      = (__bf16)xa0[i];
            a[4 + i]  = (__bf16)xa1[i];
            a[8 + i]  = (__bf16)xa2[i];
            a[12 + i] = (__bf16)xa3[i];
        }
        // issue next iteration's A loads now; they overlap the 12-WMMA chain
        if (kk + 32 < Cc) {
            xa0 = *(const v4f*)(pxbase + kk + 32);
            xa1 = *(const v4f*)(pxbase + kk + 36);
            xa2 = *(const v4f*)(pxbase + kk + 40);
            xa3 = *(const v4f*)(pxbase + kk + 44);
        }
        // ---- 12 WMMAs with depth-2 B-fragment pipeline ----
#pragma unroll
        for (int idx = 0; idx < 12; ++idx) {
            v16bf bcur = (idx & 1) ? bq1 : bq0;
            int nidx = idx + 2;                      // load 2 tiles ahead
            if (nidx < 12) {
                if (idx & 1) bq1 = ldB(nidx, kk); else bq0 = ldB(nidx, kk);
            } else if (kk + 32 < Cc) {
                if (idx & 1) bq1 = ldB(nidx - 12, kk + 32);
                else         bq0 = ldB(nidx - 12, kk + 32);
            }
            acc[idx] = __builtin_amdgcn_wmma_f32_16x16x32_bf16(
                false, a, false, bcur, (short)0, acc[idx], false, false);
        }
    }

    const int b    = row0 / Tc;
    const int tloc = row0 % Tc;
#pragma unroll
    for (int nt = 0; nt < 4; ++nt) {
#pragma unroll
        for (int r = 0; r < 8; ++r) {
            int row = row0 + r + 8 * half;           // C/D layout: lane=N, vgpr r = row
            qb[(size_t)row * Hc + nt * 16 + nl] = (__bf16)acc[0 * 4 + nt][r];
            kb[(size_t)row * Hc + nt * 16 + nl] = (__bf16)acc[1 * 4 + nt][r];
        }
        // V transposed: vt[b][h][t]; rows r are consecutive t -> one 16B store
        int h = nt * 16 + nl;
        v8bf pk;
#pragma unroll
        for (int r = 0; r < 8; ++r) pk[r] = (__bf16)acc[2 * 4 + nt][r];
        *(v8bf*)(vt + ((size_t)b * Hc + h) * Tc + tloc + 8 * half) = pk;
    }
}

// ---------------------------------------------------------------------------
// Flash-attention strip: 16 query rows per wave, key tiles of 32 strided by 64
// (phase = even/odd tiles). Writes UNNORMALIZED partial O plus (m, l).
// ---------------------------------------------------------------------------
__device__ __forceinline__ void flash_strip(
        const __bf16* __restrict__ qb, const __bf16* __restrict__ kbuf,
        const __bf16* __restrict__ vt,
        float* __restrict__ opart, float* __restrict__ mpart, float* __restrict__ lpart,
        __bf16 (*pbuf)[40],
        int b, int qrow0, int phase, int lane) {
    const int nl   = lane & 15;
    const int half = lane >> 4;

    // Q A-fragments: K-depth 64 over H -> two 16x32 fragments
    const __bf16* qrow = qb + ((size_t)b * Tc + qrow0 + nl) * Hc;
    v16bf aq[2];
#pragma unroll
    for (int j = 0; j < 2; ++j) {
        const __bf16* p = qrow + j * 32 + half * 16;
        v8bf lo = *(const v8bf*)p;
        v8bf hi = *(const v8bf*)(p + 8);
#pragma unroll
        for (int i = 0; i < 8; ++i) { aq[j][i] = lo[i]; aq[j][8 + i] = hi[i]; }
    }

    v8f o[4] = {};
    float mrow[8], lrow[8];
#pragma unroll
    for (int r = 0; r < 8; ++r) { mrow[r] = -1e30f; lrow[r] = 0.f; }

    for (int kt = phase * 32; kt < qrow0 + 16; kt += 64) {
        // prefetch next key tile of this phase (L2-resident; global_prefetch_b8)
        if (kt + 64 < qrow0 + 16) {
            __builtin_prefetch(kbuf + ((size_t)b * Tc + kt + 64 + nl) * Hc, 0, 3);
            __builtin_prefetch(vt + ((size_t)b * Hc + nl) * Tc + kt + 64, 0, 3);
        }
        // ---- S = Q K^T (two 16x16 sub-tiles of 16 keys each) ----
        v8f s[2];
#pragma unroll
        for (int s2 = 0; s2 < 2; ++s2) {
            v8f sa = {};
            const __bf16* krow = kbuf + ((size_t)b * Tc + kt + s2 * 16 + nl) * Hc;
#pragma unroll
            for (int j = 0; j < 2; ++j) {
                v16bf bk = *(const v16bf*)(krow + j * 32 + half * 16);
                sa = __builtin_amdgcn_wmma_f32_16x16x32_bf16(
                    false, aq[j], false, bk, (short)0, sa, false, false);
            }
            s[s2] = sa;
        }
        // ---- issue V B-fragment loads now: they overlap mask+softmax VALU ----
        v16bf bv[4];
#pragma unroll
        for (int t = 0; t < 4; ++t) {
            const __bf16* vrow = vt + ((size_t)b * Hc + t * 16 + nl) * Tc + kt + half * 16;
            bv[t] = *(const v16bf*)vrow;
        }
        // ---- causal mask (only for tiles straddling the diagonal) ----
        if (kt + 31 > qrow0) {
#pragma unroll
            for (int s2 = 0; s2 < 2; ++s2) {
                int key = kt + s2 * 16 + nl;
#pragma unroll
                for (int r = 0; r < 8; ++r) {
                    int row = qrow0 + r + 8 * half;
                    if (key > row) s[s2][r] = -1e30f;
                }
            }
        }
        // ---- online softmax (lanes 0-15 / 16-31 are the two row halves) ----
        float scale[8];
#pragma unroll
        for (int r = 0; r < 8; ++r) {
            float mx = fmaxf(s[0][r], s[1][r]);
#pragma unroll
            for (int off = 8; off >= 1; off >>= 1)
                mx = fmaxf(mx, __shfl_xor(mx, off, 32));
            float mnew = fmaxf(mrow[r], mx);
            float sc   = __expf(mrow[r] - mnew);
            float p0   = __expf(s[0][r] - mnew);
            float p1   = __expf(s[1][r] - mnew);
            float rs   = p0 + p1;
#pragma unroll
            for (int off = 8; off >= 1; off >>= 1)
                rs += __shfl_xor(rs, off, 32);
            lrow[r]  = lrow[r] * sc + rs;
            mrow[r]  = mnew;
            scale[r] = sc;
            s[0][r] = p0; s[1][r] = p1;
        }
#pragma unroll
        for (int t = 0; t < 4; ++t)
#pragma unroll
            for (int r = 0; r < 8; ++r) o[t][r] *= scale[r];

        // ---- P: accumulator layout -> A-fragment layout via wave-private LDS ----
#pragma unroll
        for (int s2 = 0; s2 < 2; ++s2)
#pragma unroll
            for (int r = 0; r < 8; ++r)
                pbuf[r + 8 * half][s2 * 16 + nl] = (__bf16)s[s2][r];
        // same-wave DS ops are in order; compiler inserts s_wait_dscnt
        v16bf pa;
        {
            const __bf16* pp = &pbuf[nl][half * 16];
            v8bf lo = *(const v8bf*)pp;
            v8bf hi = *(const v8bf*)(pp + 8);
#pragma unroll
            for (int i = 0; i < 8; ++i) { pa[i] = lo[i]; pa[8 + i] = hi[i]; }
        }
        // ---- O += P V  (V fragments already in registers) ----
#pragma unroll
        for (int t = 0; t < 4; ++t) {
            o[t] = __builtin_amdgcn_wmma_f32_16x16x32_bf16(
                false, pa, false, bv[t], (short)0, o[t], false, false);
        }
    }

    // ---- write partials (unnormalized) ----
    size_t obase = (size_t)phase * BTH;
#pragma unroll
    for (int t = 0; t < 4; ++t)
#pragma unroll
        for (int r = 0; r < 8; ++r) {
            int row = qrow0 + r + 8 * half;
            opart[obase + ((size_t)b * Tc + row) * Hc + t * 16 + nl] = o[t][r];
        }
    if (nl == 0) {
#pragma unroll
        for (int r = 0; r < 8; ++r) {
            int row = qrow0 + r + 8 * half;
            mpart[(size_t)phase * BT + (size_t)b * Tc + row] = mrow[r];
            lpart[(size_t)phase * BT + (size_t)b * Tc + row] = lrow[r];
        }
    }
}

// ---------------------------------------------------------------------------
// Kernel 2: causal flash attention, balanced.
//   grid = (32 strip-pairs, B, 2 key-parity phases); 4 waves / block.
//   Block processes complementary strips qt and 63-qt -> uniform work.
// ---------------------------------------------------------------------------
__global__ __launch_bounds__(128) void flash_attn_kernel(
        const __bf16* __restrict__ qb, const __bf16* __restrict__ kbuf,
        const __bf16* __restrict__ vt,
        float* __restrict__ opart, float* __restrict__ mpart, float* __restrict__ lpart) {
    __shared__ __bf16 pbuf[4][16][40];   // pad 32->40 elems: conflict-free b128 reads
    const int lane  = threadIdx.x & 31;
    const int wave  = threadIdx.x >> 5;
    const int b     = blockIdx.y;
    const int phase = blockIdx.z;

    int qtA = blockIdx.x;            // light strip
    int qtB = 63 - blockIdx.x;       // heavy strip (pairing balances the triangle)
    flash_strip(qb, kbuf, vt, opart, mpart, lpart, pbuf[wave],
                b, qtB * 64 + wave * 16, phase, lane);   // heavy first
    flash_strip(qb, kbuf, vt, opart, mpart, lpart, pbuf[wave],
                b, qtA * 64 + wave * 16, phase, lane);
}

// ---------------------------------------------------------------------------
// Kernel 3: merge the two key-phase partials:
//   out = (o0*w0 + o1*w1) / (l0*w0 + l1*w1),  wi = exp(mi - max(m0,m1))
// ---------------------------------------------------------------------------
__global__ void combine_kernel(const float* __restrict__ opart,
                               const float* __restrict__ mpart,
                               const float* __restrict__ lpart,
                               float* __restrict__ out) {
    int idx = blockIdx.x * blockDim.x + threadIdx.x;
    if (idx >= BTH) return;
    int row = idx / Hc;
    float m0 = mpart[row], m1 = mpart[BT + row];
    float l0 = lpart[row], l1 = lpart[BT + row];
    float M  = fmaxf(m0, m1);
    float w0 = __expf(m0 - M), w1 = __expf(m1 - M);
    float denom = l0 * w0 + l1 * w1;
    out[idx] = (opart[idx] * w0 + opart[(size_t)BTH + idx] * w1) / denom;
}

// ---------------------------------------------------------------------------
extern "C" void kernel_launch(void* const* d_in, const int* in_sizes, int n_in,
                              void* d_out, int out_size, void* d_ws, size_t ws_size,
                              hipStream_t stream) {
    const float* x  = (const float*)d_in[0];
    const float* Wq = (const float*)d_in[1];
    const float* Wk = (const float*)d_in[2];
    const float* Wv = (const float*)d_in[3];
    float* out = (float*)d_out;

    __bf16* ws   = (__bf16*)d_ws;
    __bf16* qb   = ws;                                 // B*T*H bf16
    __bf16* kb   = qb + (size_t)BTH;                   // B*T*H bf16
    __bf16* vt   = kb + (size_t)BTH;                   // B*H*T bf16 (transposed)
    __bf16* Wt   = vt + (size_t)BTH;                   // 3*H*C bf16
    float*  fws   = (float*)(Wt + (size_t)3 * Hc * Cc);
    float*  opart = fws;                               // [2][B*T*H] fp32
    float*  mpart = opart + (size_t)2 * BTH;           // [2][B*T] fp32
    float*  lpart = mpart + (size_t)2 * BT;            // [2][B*T] fp32

    int wtN = 3 * Cc * Hc;
    wt_transpose_kernel<<<(wtN + 255) / 256, 256, 0, stream>>>(Wq, Wk, Wv, Wt);

    qkv_proj_kernel<<<BT / 128, 256, 0, stream>>>(x, Wt, qb, kb, vt);

    dim3 grid(Tc / 128, Bc, 2);   // 32 strip-pairs x 4 batches x 2 key phases
    flash_attn_kernel<<<grid, 128, 0, stream>>>(qb, kb, vt, opart, mpart, lpart);

    combine_kernel<<<(BTH + 255) / 256, 256, 0, stream>>>(opart, mpart, lpart, out);
}
